// single_multi_head_31344671326318
// MI455X (gfx1250) — matrix operand order repaired and checked by reading, not compiled, not run on hardware
//
#include <hip/hip_runtime.h>
#include <stdint.h>

typedef __bf16 bf16;
typedef __attribute__((ext_vector_type(8)))  float v8f;
typedef __attribute__((ext_vector_type(16))) bf16  v16bf;
typedef __attribute__((ext_vector_type(8)))  bf16  v8bf;

#define N_Q   4096
#define N_K   4096
#define DIM   1024
#define BM    128
#define BN    128
#define BK    64   // K-slab for the async GEMM core (two WMMA K-steps)
#define LDSS  72   // padded LDS row stride for BK=64 tiles (144 bytes)
#define BKO   32   // K-slab for the out kernel
#define LDSO  40   // padded LDS row stride for BK=32 tiles (80 bytes)

// ---------------------------------------------------------------------------
// CDNA5 async global->LDS copy (ASYNCcnt-tracked). IOFFSET is added to BOTH
// the global address and the LDS address, so one base covers a 64B row chunk.
__device__ __forceinline__ void cp_async_row64(unsigned lds, const bf16* g) {
  asm volatile("global_load_async_to_lds_b128 %0, %1, off"
               :: "v"(lds), "v"(g) : "memory");
  asm volatile("global_load_async_to_lds_b128 %0, %1, off offset:16"
               :: "v"(lds), "v"(g) : "memory");
  asm volatile("global_load_async_to_lds_b128 %0, %1, off offset:32"
               :: "v"(lds), "v"(g) : "memory");
  asm volatile("global_load_async_to_lds_b128 %0, %1, off offset:48"
               :: "v"(lds), "v"(g) : "memory");
}

// Concatenate two 8-element bf16 vectors into a 16-element WMMA fragment.
__device__ __forceinline__ v16bf cat8(v8bf lo, v8bf hi) {
  return __builtin_shufflevector(lo, hi, 0,1,2,3,4,5,6,7,8,9,10,11,12,13,14,15);
}

// One K=32 WMMA step on staged LDS tiles starting at column kb.
// Wave computes a 32(M) x 64(N) sub-tile as a 2x4 grid of 16x16 WMMAs.
__device__ __forceinline__ void mma_step(const bf16* sA, const bf16* sB,
                                         int ldss, int kb,
                                         int waveM, int waveN, int lane,
                                         v8f acc[2][4]) {
  const int lmod  = lane & 15;
  const int lhalf = lane >> 4;
  v16bf a[2];
#pragma unroll
  for (int i = 0; i < 2; ++i) {
    const bf16* p = sA + (waveM * 32 + i * 16 + lmod) * ldss + kb + lhalf * 8;
    a[i] = cat8(*(const v8bf*)p, *(const v8bf*)(p + 16));
  }
  v16bf b[4];
#pragma unroll
  for (int j = 0; j < 4; ++j) {
    const bf16* p = sB + (waveN * 64 + j * 16 + lmod) * ldss + kb + lhalf * 8;
    b[j] = cat8(*(const v8bf*)p, *(const v8bf*)(p + 16));
  }
#pragma unroll
  for (int i = 0; i < 2; ++i)
#pragma unroll
    for (int j = 0; j < 4; ++j)
      acc[i][j] = __builtin_amdgcn_wmma_f32_16x16x32_bf16(
          false, a[i], false, b[j], (short)0, acc[i][j], false, false);
}

// Async, double-buffered bf16 GEMM core: acc += A[mBase:+128, :] * B[nBase:+128, :]^T
// (B rows are the "N" index, K-contiguous). BK=64 per buffer: per thread per
// slab 4 async b128 for A + 4 for B -> ASYNCcnt += 8 per wave per slab.
// Last iteration peeled so the steady-state loop is branch-free.
__device__ __forceinline__ void gemm_bf16_core(
    const bf16* __restrict__ A, int lda, const bf16* __restrict__ B, int ldb,
    int kdim, int mBase, int nBase, bf16* sA, bf16* sB,
    int waveM, int waveN, int lane, v8f acc[2][4]) {
  const int t   = threadIdx.x;
  const int row = t >> 1;   // 0..127
  const int kh  = t & 1;    // 32-element half of the BK=64 slab
  const bf16* gA = A + (size_t)(mBase + row) * lda + kh * 32;
  const bf16* gB = B + (size_t)(nBase + row) * ldb + kh * 32;
  const unsigned la = (unsigned)(uintptr_t)(sA + row * LDSS + kh * 32);
  const unsigned lb = (unsigned)(uintptr_t)(sB + row * LDSS + kh * 32);
  const unsigned bstep = BM * LDSS * (unsigned)sizeof(bf16);  // buf0 -> buf1 bytes
  const int tile = BM * LDSS;                                  // elements
  const int nIter = kdim / BK;

  // Prologue: fill buffer 0 with the first slab.
  cp_async_row64(la, gA);
  cp_async_row64(lb, gB);
  int buf = 0;
  for (int i = 0; i < nIter - 1; ++i) {
    // Prefetch next slab into the other buffer, retire the previous group.
    cp_async_row64(la + (buf ^ 1) * bstep, gA + (i + 1) * BK);
    cp_async_row64(lb + (buf ^ 1) * bstep, gB + (i + 1) * BK);
    asm volatile("s_wait_asynccnt 0x8" ::: "memory");
    __syncthreads();
    mma_step(sA + buf * tile, sB + buf * tile, LDSS, 0,  waveM, waveN, lane, acc);
    mma_step(sA + buf * tile, sB + buf * tile, LDSS, 32, waveM, waveN, lane, acc);
    __syncthreads();  // protect buf from next iteration's async writes
    buf ^= 1;
  }
  asm volatile("s_wait_asynccnt 0x0" ::: "memory");
  __syncthreads();
  mma_step(sA + buf * tile, sB + buf * tile, LDSS, 0,  waveM, waveN, lane, acc);
  mma_step(sA + buf * tile, sB + buf * tile, LDSS, 32, waveM, waveN, lane, acc);
}

// ---------------- Kernel 0: fp32 -> bf16 bulk convert -------------------------
__global__ __launch_bounds__(256) void cvt_kernel(const float* __restrict__ src,
                                                  bf16* __restrict__ dst) {
  const size_t i = ((size_t)blockIdx.x * 256 + threadIdx.x) * 16;
  const float4* s4 = (const float4*)(src + i);
  float4 f0 = s4[0], f1 = s4[1], f2 = s4[2], f3 = s4[3];
  v8bf a, b;
  a[0] = (bf16)f0.x; a[1] = (bf16)f0.y; a[2] = (bf16)f0.z; a[3] = (bf16)f0.w;
  a[4] = (bf16)f1.x; a[5] = (bf16)f1.y; a[6] = (bf16)f1.z; a[7] = (bf16)f1.w;
  b[0] = (bf16)f2.x; b[1] = (bf16)f2.y; b[2] = (bf16)f2.z; b[3] = (bf16)f2.w;
  b[4] = (bf16)f3.x; b[5] = (bf16)f3.y; b[6] = (bf16)f3.z; b[7] = (bf16)f3.w;
  *(v8bf*)(dst + i)     = a;
  *(v8bf*)(dst + i + 8) = b;
}

// ---------------- Kernel 1: Y = X @ W^T + b  (bf16 in, bf16 out) --------------
__global__ __launch_bounds__(256) void proj_kernel(
    const bf16* __restrict__ qc, const bf16* __restrict__ kc,
    const bf16* __restrict__ vc,
    const bf16* __restrict__ Wqc, const float* __restrict__ bq,
    const bf16* __restrict__ Wkc, const float* __restrict__ bk,
    const bf16* __restrict__ Wvc, const float* __restrict__ bv,
    bf16* __restrict__ Qb, bf16* __restrict__ Kb, bf16* __restrict__ Vb) {
  const bf16* X; const bf16* W; const float* bias; bf16* Y;
  if (blockIdx.z == 0)      { X = qc; W = Wqc; bias = bq; Y = Qb; }
  else if (blockIdx.z == 1) { X = kc; W = Wkc; bias = bk; Y = Kb; }
  else                      { X = vc; W = Wvc; bias = bv; Y = Vb; }

  __shared__ __attribute__((aligned(16))) bf16 sA[2 * BM * LDSS];
  __shared__ __attribute__((aligned(16))) bf16 sB[2 * BN * LDSS];

  const int t     = threadIdx.x;
  const int lane  = t & 31;
  const int wave  = t >> 5;
  const int waveM = wave & 3;
  const int waveN = wave >> 2;
  const int mBase = blockIdx.y * BM;
  const int nBase = blockIdx.x * BN;

  v8f acc[2][4] = {};
  gemm_bf16_core(X, DIM, W, DIM, DIM, mBase, nBase, sA, sB, waveM, waveN, lane, acc);

  const int lmod  = lane & 15;
  const int lhalf = lane >> 4;
#pragma unroll
  for (int j = 0; j < 4; ++j) {
    const int gn = nBase + waveN * 64 + j * 16 + lmod;
    const float bj = bias[gn];
#pragma unroll
    for (int i = 0; i < 2; ++i) {
      const int gm0 = mBase + waveM * 32 + i * 16 + lhalf * 8;
#pragma unroll
      for (int r = 0; r < 8; ++r)
        Y[(size_t)(gm0 + r) * DIM + gn] = (bf16)(acc[i][j][r] + bj);
    }
  }
}

// ---------------- Kernel 2: S = (Q @ K^T) / N_K  (bf16 in, fp32 out) ----------
__global__ __launch_bounds__(256) void scores_kernel(
    const bf16* __restrict__ Qb, const bf16* __restrict__ Kb,
    float* __restrict__ S) {
  __shared__ __attribute__((aligned(16))) bf16 sA[2 * BM * LDSS];
  __shared__ __attribute__((aligned(16))) bf16 sB[2 * BN * LDSS];

  const int t     = threadIdx.x;
  const int lane  = t & 31;
  const int wave  = t >> 5;
  const int waveM = wave & 3;
  const int waveN = wave >> 2;
  const int mBase = blockIdx.y * BM;   // query rows
  const int nBase = blockIdx.x * BN;   // key rows

  v8f acc[2][4] = {};
  gemm_bf16_core(Qb, DIM, Kb, DIM, DIM, mBase, nBase, sA, sB, waveM, waveN, lane, acc);

  const int lmod  = lane & 15;
  const int lhalf = lane >> 4;
  const float scale = 1.0f / (float)N_K;
#pragma unroll
  for (int j = 0; j < 4; ++j) {
    const int gn = nBase + waveN * 64 + j * 16 + lmod;
#pragma unroll
    for (int i = 0; i < 2; ++i) {
      const int gm0 = mBase + waveM * 32 + i * 16 + lhalf * 8;
#pragma unroll
      for (int r = 0; r < 8; ++r)
        S[(size_t)(gm0 + r) * N_K + gn] = acc[i][j][r] * scale;
    }
  }
}

// ---------------- Kernel 3: column-wise softmax statistics --------------------
__global__ __launch_bounds__(256) void colreduce_kernel(
    const float* __restrict__ S, float* __restrict__ cmax,
    float* __restrict__ crinv) {
  __shared__ float red[4][64];
  const int tx = threadIdx.x & 63;
  const int ty = threadIdx.x >> 6;
  const int j  = blockIdx.x * 64 + tx;

  float m = -3.4e38f;
  for (int i = ty; i < N_Q; i += 4)
    m = fmaxf(m, S[(size_t)i * N_K + j]);
  red[ty][tx] = m;
  __syncthreads();
  if (ty == 0) {
    m = fmaxf(fmaxf(red[0][tx], red[1][tx]), fmaxf(red[2][tx], red[3][tx]));
    red[0][tx] = m;
  }
  __syncthreads();
  m = red[0][tx];

  float s = 0.0f;
  for (int i = ty; i < N_Q; i += 4)
    s += __expf(S[(size_t)i * N_K + j] - m);
  __syncthreads();
  red[ty][tx] = s;
  __syncthreads();
  if (ty == 0) {
    s = red[0][tx] + red[1][tx] + red[2][tx] + red[3][tx];
    cmax[j]  = m;
    crinv[j] = 1.0f / s;
  }
}

// ---------------- Kernel 4: out = softmax(S) @ V ------------------------------
// Register-staged, double-buffered pipeline with ONE barrier per K-step:
//   load regs(k+1) | barrier | mma(buf) | stage regs -> buf^1
// Softmax normalization fused into A staging; V transposed into LDS.
__global__ __launch_bounds__(256) void out_kernel(
    const float* __restrict__ S, const float* __restrict__ cmax,
    const float* __restrict__ crinv, const bf16* __restrict__ Vb,
    float* __restrict__ out) {
  __shared__ __attribute__((aligned(16))) bf16 sA[2 * BM * LDSO];
  __shared__ __attribute__((aligned(16))) bf16 sB[2 * BN * LDSO];

  const int t     = threadIdx.x;
  const int lane  = t & 31;
  const int wave  = t >> 5;
  const int waveM = wave & 3;
  const int waveN = wave >> 2;
  const int mBase = blockIdx.y * BM;   // query rows
  const int nBase = blockIdx.x * BN;   // feature cols of V
  const int row   = t >> 1;
  const int kh    = t & 1;
  const int jrow  = t & 31;            // B staging: V row within BK slab
  const int dgrp  = (t >> 5) * 16;     // B staging: 16-wide feature group
  const int tile  = BM * LDSO;         // elements per buffer

  float4 sv[4], mv[4], rv[4];
  v8bf   vv0, vv1;

  auto load_regs = [&](int k0) {
    const int j0 = k0 + kh * 16;
    const float4* s4 = (const float4*)(S + (size_t)(mBase + row) * N_K + j0);
    const float4* m4 = (const float4*)(cmax + j0);
    const float4* r4 = (const float4*)(crinv + j0);
#pragma unroll
    for (int g = 0; g < 4; ++g) { sv[g] = s4[g]; mv[g] = m4[g]; rv[g] = r4[g]; }
    const v8bf* s8 = (const v8bf*)(Vb + (size_t)(k0 + jrow) * DIM + nBase + dgrp);
    vv0 = s8[0]; vv1 = s8[1];
  };
  auto stage = [&](bf16* dA, bf16* dB) {
    bf16* d = dA + row * LDSO + kh * 16;
#pragma unroll
    for (int g = 0; g < 4; ++g) {
      d[4 * g + 0] = (bf16)(__expf(sv[g].x - mv[g].x) * rv[g].x);
      d[4 * g + 1] = (bf16)(__expf(sv[g].y - mv[g].y) * rv[g].y);
      d[4 * g + 2] = (bf16)(__expf(sv[g].z - mv[g].z) * rv[g].z);
      d[4 * g + 3] = (bf16)(__expf(sv[g].w - mv[g].w) * rv[g].w);
    }
#pragma unroll
    for (int e = 0; e < 8; ++e) {
      dB[(dgrp + e) * LDSO + jrow]     = vv0[e];
      dB[(dgrp + 8 + e) * LDSO + jrow] = vv1[e];
    }
  };

  v8f acc[2][4] = {};
  load_regs(0);
  stage(sA, sB);
  int buf = 0;
  for (int k0 = 0; k0 < N_K; k0 += BKO) {
    const bool next = (k0 + BKO) < N_K;
    if (next) load_regs(k0 + BKO);   // global latency overlaps barrier + MMA
    __syncthreads();                 // current buffer fully staged by all waves
    mma_step(sA + buf * tile, sB + buf * tile, LDSO, 0, waveM, waveN, lane, acc);
    if (next) stage(sA + (buf ^ 1) * tile, sB + (buf ^ 1) * tile);
    buf ^= 1;
  }

  const int lmod  = lane & 15;
  const int lhalf = lane >> 4;
#pragma unroll
  for (int j = 0; j < 4; ++j) {
    const int gn = nBase + waveN * 64 + j * 16 + lmod;
#pragma unroll
    for (int i = 0; i < 2; ++i) {
      const int gm0 = mBase + waveM * 32 + i * 16 + lhalf * 8;
#pragma unroll
      for (int r = 0; r < 8; ++r)
        out[(size_t)(gm0 + r) * DIM + gn] = acc[i][j][r];
    }
  }
}

// ---------------- Host launcher ----------------------------------------------
extern "C" void kernel_launch(void* const* d_in, const int* in_sizes, int n_in,
                              void* d_out, int out_size, void* d_ws, size_t ws_size,
                              hipStream_t stream) {
  const float* q  = (const float*)d_in[0];
  const float* k  = (const float*)d_in[1];
  const float* v  = (const float*)d_in[2];
  const float* Wq = (const float*)d_in[3];
  const float* bq = (const float*)d_in[4];
  const float* Wk = (const float*)d_in[5];
  const float* bk = (const float*)d_in[6];
  const float* Wv = (const float*)d_in[7];
  const float* bv = (const float*)d_in[8];

  char* ws = (char*)d_ws;
  const size_t MB = 1024 * 1024;
  // Workspace layout (bytes):
  //   [0,8M) qc  [8M,16M) kc  [16M,24M) vc          (bf16 copies of inputs)
  //   [24M,26M) Wqc [26M,28M) Wkc [28M,30M) Wvc     (bf16 copies of weights)
  //   [30M,38M) Qb  [38M,46M) Kb  [46M,54M) Vb      (projected, bf16)
  //   [54M,118M) S fp32 (4096x4096)
  //   [118M,+16K) colmax   [+16K,+32K) colrinv
  bf16*  qc    = (bf16*)(ws);
  bf16*  kc    = (bf16*)(ws + 8 * MB);
  bf16*  vc    = (bf16*)(ws + 16 * MB);
  bf16*  Wqc   = (bf16*)(ws + 24 * MB);
  bf16*  Wkc   = (bf16*)(ws + 26 * MB);
  bf16*  Wvc   = (bf16*)(ws + 28 * MB);
  bf16*  Qb    = (bf16*)(ws + 30 * MB);
  bf16*  Kb    = (bf16*)(ws + 38 * MB);
  bf16*  Vb    = (bf16*)(ws + 46 * MB);
  float* S     = (float*)(ws + 54 * MB);
  float* cmax  = (float*)(ws + 118 * MB);
  float* crinv = cmax + N_K;

  dim3 blk(256);
  const int big = (N_Q * DIM) / (256 * 16);    // 1024 blocks for 4M elements
  const int wsm = (DIM * DIM) / (256 * 16);    // 256 blocks for 1M elements
  cvt_kernel<<<big, blk, 0, stream>>>(q, qc);
  cvt_kernel<<<big, blk, 0, stream>>>(k, kc);
  cvt_kernel<<<big, blk, 0, stream>>>(v, vc);
  cvt_kernel<<<wsm, blk, 0, stream>>>(Wq, Wqc);
  cvt_kernel<<<wsm, blk, 0, stream>>>(Wk, Wkc);
  cvt_kernel<<<wsm, blk, 0, stream>>>(Wv, Wvc);

  proj_kernel<<<dim3(DIM / BN, N_Q / BM, 3), blk, 0, stream>>>(
      qc, kc, vc, Wqc, bq, Wkc, bk, Wvc, bv, Qb, Kb, Vb);
  scores_kernel<<<dim3(N_K / BN, N_Q / BM), blk, 0, stream>>>(Qb, Kb, S);
  colreduce_kernel<<<dim3(N_K / 64), blk, 0, stream>>>(S, cmax, crinv);
  out_kernel<<<dim3(DIM / BN, N_Q / BM), blk, 0, stream>>>(
      S, cmax, crinv, Vb, (float*)d_out);
}
